// GcnEncoder_33277406609923
// MI455X (gfx1250) — matrix operand order repaired and checked by reading, not compiled, not run on hardware
//
#include <hip/hip_runtime.h>
#include <hip/hip_bf16.h>
#include <math.h>

typedef __bf16 v16bf __attribute__((ext_vector_type(16)));
typedef float  v8f   __attribute__((ext_vector_type(8)));
typedef unsigned int uint4v __attribute__((ext_vector_type(4)));

#define RANK 256
#define EPSV 1e-5f

// ---------------------------------------------------------------------------
// helpers
// ---------------------------------------------------------------------------
__device__ __forceinline__ void atomicMaxFloat(float* addr, float v) {
    // sign-split trick: monotone under concurrent mixed-sign updates,
    // init value must be -inf (uint 0xFF800000 / int large-negative).
    if (v >= 0.0f) atomicMax((int*)addr, __float_as_int(v));
    else           atomicMin((unsigned int*)addr, __float_as_uint(v));
}

// ---------------------------------------------------------------------------
// degree / normalization
// ---------------------------------------------------------------------------
__global__ void k_init_deg(float* deg, int n) {
    int i = blockIdx.x * blockDim.x + threadIdx.x;
    if (i < n) deg[i] = 1.0f;                 // self loop
}
__global__ void k_count_deg(const int* __restrict__ dst, float* deg, int e) {
    int i = blockIdx.x * blockDim.x + threadIdx.x;
    if (i < e) atomicAdd(&deg[dst[i]], 1.0f);
}
__global__ void k_rsqrt_deg(float* deg, int n) {
    int i = blockIdx.x * blockDim.x + threadIdx.x;
    if (i < n) deg[i] = rsqrtf(deg[i]);       // deg >= 1 always
}

// ---------------------------------------------------------------------------
// precision conversion / weight packing
// ---------------------------------------------------------------------------
__global__ void k_f32_to_bf16(const float* __restrict__ src, __bf16* __restrict__ dst, int n) {
    int i = blockIdx.x * blockDim.x + threadIdx.x;
    if (i < n) dst[i] = (__bf16)src[i];
}
// Wp[p][c] packs W[2p][c] (lo) | W[2p+1][c] (hi)  -> single b32 B-fragment loads
__global__ void k_pack_w(const float* __restrict__ W, unsigned* __restrict__ Wp, int halfK) {
    int i = blockIdx.x * blockDim.x + threadIdx.x;
    if (i >= halfK * RANK) return;
    int p = i / RANK, c = i % RANK;
    union { __bf16 h[2]; unsigned u; } pk;
    pk.h[0] = (__bf16)W[(size_t)(2 * p    ) * RANK + c];
    pk.h[1] = (__bf16)W[(size_t)(2 * p + 1) * RANK + c];
    Wp[i] = pk.u;
}

// ---------------------------------------------------------------------------
// GEMM: C[M,256] = A[M,K](bf16) x W[K,256].
// One wave computes a 16x64 tile (4 f32 accumulators) -> 4 WMMAs per
// A-fragment fetch; block = 4 waves = full 256 columns of a 16-row slab.
// A fragment in contiguous K order (k = 16*hi + s): lane m -> row r0+m,
// lanes 0-15 K=0..15, lanes 16-31 K=16..31; two b128 loads.
// B fragments come from K-pair-packed W; all displacements constant so the
// 32 b32 loads fold into immediate offsets off one base.
// ---------------------------------------------------------------------------
__global__ __launch_bounds__(128) void k_gemm_bf16_wmma(
    const __bf16* __restrict__ A, const unsigned* __restrict__ Bp,
    float* __restrict__ C, int M, int K) {
    const int lane = threadIdx.x & 31;
    const int wave = threadIdx.x >> 5;         // 0..3 -> 64-column slab
    const int r0 = blockIdx.x << 4;
    const int c0 = wave << 6;
    const int m  = lane & 15;
    const int hi = lane >> 4;
    const unsigned* arow  = (const unsigned*)(A + (size_t)(r0 + m) * K) + (hi ? 8 : 0);
    const unsigned* bbase = Bp + (size_t)(hi ? 8 : 0) * RANK + c0 + m;
    v8f acc0 = {}, acc1 = {}, acc2 = {}, acc3 = {};
    for (int kt = 0; kt < (K >> 5); ++kt) {
        union { v16bf v; uint4v q[2]; } a;
        a.q[0] = *(const uint4v*)(arow + kt * 16);
        a.q[1] = *(const uint4v*)(arow + kt * 16 + 4);
        __builtin_prefetch(arow + kt * 16 + 16, 0, 1);   // next K-step A row
        const unsigned* bb = bbase + (size_t)kt * 16 * RANK;
        union { v16bf v; unsigned u[8]; } b0, b1, b2, b3;
#pragma unroll
        for (int v = 0; v < 8; ++v) {
            b0.u[v] = bb[v * RANK];
            b1.u[v] = bb[v * RANK + 16];
            b2.u[v] = bb[v * RANK + 32];
            b3.u[v] = bb[v * RANK + 48];
        }
        acc0 = __builtin_amdgcn_wmma_f32_16x16x32_bf16(false, a.v, false, b0.v, (short)0, acc0, false, false);
        acc1 = __builtin_amdgcn_wmma_f32_16x16x32_bf16(false, a.v, false, b1.v, (short)0, acc1, false, false);
        acc2 = __builtin_amdgcn_wmma_f32_16x16x32_bf16(false, a.v, false, b2.v, (short)0, acc2, false, false);
        acc3 = __builtin_amdgcn_wmma_f32_16x16x32_bf16(false, a.v, false, b3.v, (short)0, acc3, false, false);
    }
    // D layout: VGPR j -> row r0 + hi*8 + j, col c0 + m (+16/32/48 per accum)
    float* crow = C + (size_t)(r0 + (hi << 3)) * RANK + c0 + m;
#pragma unroll
    for (int j = 0; j < 8; ++j) {
        crow[(size_t)j * RANK]      = acc0[j];
        crow[(size_t)j * RANK + 16] = acc1[j];
        crow[(size_t)j * RANK + 32] = acc2[j];
        crow[(size_t)j * RANK + 48] = acc3[j];
    }
}

// ---------------------------------------------------------------------------
// aggregation
// ---------------------------------------------------------------------------
__global__ void k_self_init(const float* __restrict__ hw, const float* __restrict__ dinv,
                            float* __restrict__ agg, int n) {
    int i = blockIdx.x * blockDim.x + threadIdx.x;
    if (i >= n * RANK) return;
    float di = dinv[i >> 8];
    agg[i] = di * di * hw[i];
}
__global__ __launch_bounds__(256) void k_edge_agg(
    const int* __restrict__ src, const int* __restrict__ dst,
    const float* __restrict__ dinv, const float* __restrict__ hw,
    float* __restrict__ agg, int E) {
    int e = blockIdx.x * 8 + (threadIdx.x >> 5);   // one wave per edge
    if (e >= E) return;
    int lane = threadIdx.x & 31;
    int s = src[e], d = dst[e];
    float nrm = dinv[s] * dinv[d];
    const float* hs = hw + (size_t)s * RANK;
    float* ad = agg + (size_t)d * RANK;
#pragma unroll
    for (int c = lane; c < RANK; c += 32)
        atomicAdd(&ad[c], nrm * hs[c]);
}

// ---------------------------------------------------------------------------
// tanh + batchnorm stats / finalize + pool
// ---------------------------------------------------------------------------
__global__ void k_zero_stats(float* sum, float* sumsq) {
    sum[threadIdx.x] = 0.0f; sumsq[threadIdx.x] = 0.0f;
}
__global__ __launch_bounds__(256) void k_tanh_stats(
    const float* __restrict__ agg, const float* __restrict__ bias,
    float* __restrict__ ybuf, float* __restrict__ sum, float* __restrict__ sumsq, int n) {
    const int c = threadIdx.x;                 // thread = channel -> coalesced
    const float b = bias[c];
    int r0 = blockIdx.x * 64;
    int r1 = min(r0 + 64, n);
    float s = 0.0f, s2 = 0.0f;
    for (int r = r0; r < r1; ++r) {
        float y = tanhf(agg[(size_t)r * RANK + c] + b);
        ybuf[(size_t)r * RANK + c] = y;
        s += y; s2 += y * y;
    }
    atomicAdd(&sum[c], s);
    atomicAdd(&sumsq[c], s2);
}
__global__ void k_bn_final(const float* sum, const float* sumsq,
                           const float* gamma, const float* beta,
                           float* scale, float* shift, float invN) {
    int c = threadIdx.x;
    float mu  = sum[c] * invN;
    float var = sumsq[c] * invN - mu * mu;     // biased var, matches jnp.var
    float sc  = gamma[c] * rsqrtf(var + EPSV);
    scale[c] = sc;
    shift[c] = beta[c] - mu * sc;
}
__global__ void k_init_pool(float* pool, int n) {
    int i = blockIdx.x * blockDim.x + threadIdx.x;
    if (i < n) pool[i] = -INFINITY;
}
__global__ void k_norm_pool(const float* __restrict__ ybuf,
                            const float* __restrict__ scale, const float* __restrict__ shift,
                            const int* __restrict__ batch,
                            float* __restrict__ pool, __bf16* __restrict__ hb, int n) {
    int i = blockIdx.x * blockDim.x + threadIdx.x;
    if (i >= n * RANK) return;
    int c = i & (RANK - 1);
    float h = scale[c] * ybuf[i] + shift[c];
    hb[i] = (__bf16)h;                          // next layer's bf16 input
    atomicMaxFloat(&pool[(size_t)batch[i >> 8] * RANK + c], h);
}
__global__ void k_pool_fix(float* pool, int n) {
    int i = blockIdx.x * blockDim.x + threadIdx.x;
    if (i < n) { float v = pool[i]; if (!isfinite(v)) pool[i] = 0.0f; }
}

// ---------------------------------------------------------------------------
// host orchestration
// ---------------------------------------------------------------------------
static inline size_t alignup(size_t x) { return (x + 255) & ~size_t(255); }

extern "C" void kernel_launch(void* const* d_in, const int* in_sizes, int n_in,
                              void* d_out, int out_size, void* d_ws, size_t ws_size,
                              hipStream_t stream) {
    const float* x     = (const float*)d_in[0];
    const int*   eidx  = (const int*)d_in[1];
    const int*   batch = (const int*)d_in[2];

    const int N   = in_sizes[2];
    const int E   = in_sizes[1] / 2;
    const int DIN = in_sizes[0] / N;
    const int Gp  = out_size / (3 * RANK);      // pooled rows per layer output
    const int* src = eidx;
    const int* dst = eidx + E;

    const float* W[3]  = {(const float*)d_in[3],  (const float*)d_in[7],  (const float*)d_in[11]};
    const float* bi[3] = {(const float*)d_in[4],  (const float*)d_in[8],  (const float*)d_in[12]};
    const float* ga[3] = {(const float*)d_in[5],  (const float*)d_in[9],  (const float*)d_in[13]};
    const float* be[3] = {(const float*)d_in[6],  (const float*)d_in[10], (const float*)d_in[14]};
    const int Kdim[3]  = {DIN, RANK, RANK};

    // workspace carve
    char* w = (char*)d_ws;
    float*    deg  = (float*)w;    w += alignup((size_t)N * 4);
    __bf16*   hb   = (__bf16*)w;   w += alignup((size_t)N * RANK * 2);
    unsigned* Wp   = (unsigned*)w; w += alignup((size_t)RANK / 2 * RANK * 4);
    float*    hw   = (float*)w;    w += alignup((size_t)N * RANK * 4);   // also ybuf
    float*    agg  = (float*)w;    w += alignup((size_t)N * RANK * 4);
    float*    sum   = (float*)w;   w += alignup(RANK * 4);
    float*    sumsq = (float*)w;   w += alignup(RANK * 4);
    float*    scale = (float*)w;   w += alignup(RANK * 4);
    float*    shift = (float*)w;   w += alignup(RANK * 4);

    const int T = 256;
    const int NC = N * RANK;

    // degree -> d^{-1/2}, once
    k_init_deg<<<(N + T - 1) / T, T, 0, stream>>>(deg, N);
    k_count_deg<<<(E + T - 1) / T, T, 0, stream>>>(dst, deg, E);
    k_rsqrt_deg<<<(N + T - 1) / T, T, 0, stream>>>(deg, N);

    // layer 0 input -> bf16
    k_f32_to_bf16<<<((size_t)N * DIN + T - 1) / T, T, 0, stream>>>(x, hb, N * DIN);

    for (int l = 0; l < 3; ++l) {
        const int K = Kdim[l];
        // pack weights (bf16, K-pair packed)
        k_pack_w<<<((K / 2) * RANK + T - 1) / T, T, 0, stream>>>(W[l], Wp, K / 2);
        // hw = h @ W  (WMMA bf16 -> f32), 4 waves x 16x64 tiles per block
        k_gemm_bf16_wmma<<<dim3(N / 16), 128, 0, stream>>>(hb, Wp, hw, N, K);
        // agg = self loop + normalized edge scatter
        k_self_init<<<(NC + T - 1) / T, T, 0, stream>>>(hw, deg, agg, N);
        k_edge_agg<<<(E + 7) / 8, 256, 0, stream>>>(src, dst, deg, hw, agg, E);
        // y = tanh(agg + b), batch stats
        k_zero_stats<<<1, RANK, 0, stream>>>(sum, sumsq);
        k_tanh_stats<<<(N + 63) / 64, RANK, 0, stream>>>(agg, bi[l], hw, sum, sumsq, N);
        k_bn_final<<<1, RANK, 0, stream>>>(sum, sumsq, ga[l], be[l], scale, shift, 1.0f / (float)N);
        // normalize, emit next bf16 input, global max pool
        float* pool = (float*)d_out + (size_t)l * Gp * RANK;
        k_init_pool<<<(Gp * RANK + T - 1) / T, T, 0, stream>>>(pool, Gp * RANK);
        k_norm_pool<<<(NC + T - 1) / T, T, 0, stream>>>(hw, scale, shift, batch, pool, hb, N);
        k_pool_fix<<<(Gp * RANK + T - 1) / T, T, 0, stream>>>(pool, Gp * RANK);
    }
}
